// MetabolismProcessor_39779987096431
// MI455X (gfx1250) — matrix operand order repaired and checked by reading, not compiled, not run on hardware
//
#include <hip/hip_runtime.h>
#include <hip/hip_bf16.h>
#include <math.h>

// ---------------------------------------------------------------------------
// MI455X (gfx1250) implementation.
// Dense 128x128 linear layers are batched across sets into big GEMMs on
// v_wmma_f32_16x16x32_bf16 (f32 accumulate). Activations are down-converted
// to bf16 once per GEMM group; weights are pre-packed into the per-lane WMMA
// B-fragment layout so the GEMM inner loop is pure b128 loads + wmma.
// Epilogue variants are compile-time (branch-free).
// ---------------------------------------------------------------------------

typedef unsigned short v8us  __attribute__((ext_vector_type(8)));
typedef unsigned short v16us __attribute__((ext_vector_type(16)));
typedef __bf16         v16bf __attribute__((ext_vector_type(16)));
typedef float          v8f   __attribute__((ext_vector_type(8)));

#define EP_RELU      1
#define EP_RES       2
#define EP_RES_BCAST 4

__device__ __forceinline__ unsigned short f2bf_bits(float f) {
  unsigned u = __builtin_bit_cast(unsigned, f);
  unsigned r = u + 0x7FFFu + ((u >> 16) & 1u);   // round-to-nearest-even
  return (unsigned short)(r >> 16);
}

// fp32 -> bf16, 8 elements per thread (packed-pair converts, b128 store).
__global__ void cvt_bf16x8(const float* __restrict__ X, unsigned short* __restrict__ Y,
                           int n8) {
  int t = blockIdx.x * blockDim.x + threadIdx.x;
  if (t >= n8) return;
  size_t base = (size_t)t * 8;
  float4 a = *(const float4*)(X + base);
  float4 b = *(const float4*)(X + base + 4);
  unsigned d0 = (unsigned)f2bf_bits(a.x) | ((unsigned)f2bf_bits(a.y) << 16);
  unsigned d1 = (unsigned)f2bf_bits(a.z) | ((unsigned)f2bf_bits(a.w) << 16);
  unsigned d2 = (unsigned)f2bf_bits(b.x) | ((unsigned)f2bf_bits(b.y) << 16);
  unsigned d3 = (unsigned)f2bf_bits(b.z) | ((unsigned)f2bf_bits(b.w) << 16);
  *(uint4*)(Y + base) = make_uint4(d0, d1, d2, d3);
}

// Pack W[K,128] (fp32) into per-lane bf16 B-fragments:
// Wp[((c*8 + tn)*32 + lane)*16 + i]  with  p=i/2,
//   kb = c*32 + ((i>>3)<<4) + ((lane>>4)<<3) + (i&7),  col = tn*16 + (lane&15)
__global__ void pack_w(const float* __restrict__ W, unsigned short* __restrict__ Wp,
                       int K) {
  int t = blockIdx.x * blockDim.x + threadIdx.x;
  if (t >= K * 128) return;
  int i  = t & 15;
  int l  = (t >> 4) & 31;
  int tn = (t >> 9) & 7;
  int c  = t >> 12;
  int p  = i >> 1;
  int kb = c * 32 + ((p >> 2) << 4) + ((l >> 4) << 3) + (i & 7);
  int col = tn * 16 + (l & 15);
  Wp[t] = f2bf_bits(W[(size_t)kb * 128 + col]);
}

// C[M,128] = epilogue(A[M,K] @ W[K,128]).  One wave per 16x16 tile.
// grid = (8, M/16), block = 32 (wave32, EXEC all ones).  KC = K/32.
// FLAGS is compile-time -> straight-line epilogue, clauseable residual loads.
template <int KC, int FLAGS>
__global__ __launch_bounds__(32) void gemm128(
    const unsigned short* __restrict__ Abf,   // bf16 row-major [M,K]
    const unsigned short* __restrict__ Wp,    // packed bf16 fragments
    const float* __restrict__ bias,
    const float* __restrict__ res,
    float* __restrict__ C, int M) {
  const int lane  = threadIdx.x;
  const int tn    = blockIdx.x;               // 0..7  (adjacent blocks share A)
  const int tm    = blockIdx.y;
  const int l15   = lane & 15;
  const int khalf = lane >> 4;
  const int row   = tm * 16 + l15;
  const int colN  = tn * 16 + l15;
  const unsigned short* arow = Abf + (size_t)row * (KC * 32);

  v8f acc = {0.f, 0.f, 0.f, 0.f, 0.f, 0.f, 0.f, 0.f};
#pragma unroll
  for (int c = 0; c < KC; ++c) {
    const int kk = c * 32;
    // A 16x32 bf16 fragment: lane<16 -> K {0..7,16..23}; lane>=16 -> {8..15,24..31}
    v8us a0 = *(const v8us*)(arow + kk + khalf * 8);        // 16B contiguous
    v8us a1 = *(const v8us*)(arow + kk + 16 + khalf * 8);   // 16B contiguous
    v16us au = __builtin_shufflevector(a0, a1, 0, 1, 2, 3, 4, 5, 6, 7,
                                       8, 9, 10, 11, 12, 13, 14, 15);
    // B fragment: 32B contiguous per lane from the packed weight image
    v16us bu = *(const v16us*)(Wp + ((size_t)c * 8 + tn) * 512 + lane * 16);
    acc = __builtin_amdgcn_wmma_f32_16x16x32_bf16(
        false, __builtin_bit_cast(v16bf, au),
        false, __builtin_bit_cast(v16bf, bu),
        (short)0, acc, false, false);
  }
  const float bb = bias ? bias[colN] : 0.f;
  const int m0 = tm * 16 + (khalf << 3);      // C layout: vgpr r -> M = m0 + r
  float rv[8];
  if (FLAGS & EP_RES) {
#pragma unroll
    for (int r = 0; r < 8; ++r)
      rv[r] = (FLAGS & EP_RES_BCAST) ? res[colN]
                                     : res[(size_t)(m0 + r) * 128 + colN];
  }
#pragma unroll
  for (int r = 0; r < 8; ++r) {
    float v = acc[r] + bb;
    if (FLAGS & EP_RELU) v = fmaxf(v, 0.f);
    if (FLAGS & EP_RES) v += rv[r];
    C[(size_t)(m0 + r) * 128 + colN] = v;
  }
}

// Self-attention within a set: thread per (set, query row, head). heads=4, d=32.
template <int N>
__global__ void attn_self(const float* __restrict__ Q, const float* __restrict__ Kb,
                          const float* __restrict__ Vb, float* __restrict__ O, int S) {
  int t = blockIdx.x * blockDim.x + threadIdx.x;
  if (t >= S * N * 4) return;
  int h = t & 3, i = (t >> 2) % N, s = t / (4 * N);
  const float* q = Q + (size_t)(s * N + i) * 128 + h * 32;
  float qr[32];
#pragma unroll
  for (int c = 0; c < 32; ++c) qr[c] = q[c];
  float sc[N], mx = -1e30f;
#pragma unroll
  for (int j = 0; j < N; ++j) {
    const float* kr = Kb + (size_t)(s * N + j) * 128 + h * 32;
    float d = 0.f;
#pragma unroll
    for (int c = 0; c < 32; ++c) d += qr[c] * kr[c];
    d *= 0.17677669529663687f;            // 1/sqrt(32)
    sc[j] = d; mx = fmaxf(mx, d);
  }
  float sum = 0.f;
#pragma unroll
  for (int j = 0; j < N; ++j) { sc[j] = __expf(sc[j] - mx); sum += sc[j]; }
  float inv = 1.f / sum, out[32];
#pragma unroll
  for (int c = 0; c < 32; ++c) out[c] = 0.f;
#pragma unroll
  for (int j = 0; j < N; ++j) {
    const float* vr = Vb + (size_t)(s * N + j) * 128 + h * 32;
    float a = sc[j] * inv;
#pragma unroll
    for (int c = 0; c < 32; ++c) out[c] += a * vr[c];
  }
  float* o = O + (size_t)(s * N + i) * 128 + h * 32;
#pragma unroll
  for (int c = 0; c < 32; ++c) o[c] = out[c];
}

// PMA attention: single shared query vector qs[128]; thread per (set, head).
template <int N>
__global__ void attn_pma(const float* __restrict__ qs, const float* __restrict__ Kb,
                         const float* __restrict__ Vb, float* __restrict__ O, int S) {
  int t = blockIdx.x * blockDim.x + threadIdx.x;
  if (t >= S * 4) return;
  int h = t & 3, s = t >> 2;
  const float* q = qs + h * 32;
  float qr[32];
#pragma unroll
  for (int c = 0; c < 32; ++c) qr[c] = q[c];
  float sc[N], mx = -1e30f;
#pragma unroll
  for (int j = 0; j < N; ++j) {
    const float* kr = Kb + (size_t)(s * N + j) * 128 + h * 32;
    float d = 0.f;
#pragma unroll
    for (int c = 0; c < 32; ++c) d += qr[c] * kr[c];
    d *= 0.17677669529663687f;
    sc[j] = d; mx = fmaxf(mx, d);
  }
  float sum = 0.f;
#pragma unroll
  for (int j = 0; j < N; ++j) { sc[j] = __expf(sc[j] - mx); sum += sc[j]; }
  float inv = 1.f / sum, out[32];
#pragma unroll
  for (int c = 0; c < 32; ++c) out[c] = 0.f;
#pragma unroll
  for (int j = 0; j < N; ++j) {
    const float* vr = Vb + (size_t)(s * N + j) * 128 + h * 32;
    float a = sc[j] * inv;
#pragma unroll
    for (int c = 0; c < 32; ++c) out[c] += a * vr[c];
  }
  float* o = O + (size_t)s * 128 + h * 32;
#pragma unroll
  for (int c = 0; c < 32; ++c) o[c] = out[c];
}

// qs = seed[1,128] @ wq + bq  (one tiny block)
__global__ void seed_q(const float* __restrict__ seed, const float* __restrict__ wq,
                       const float* __restrict__ bq, float* __restrict__ qs) {
  int o = threadIdx.x;                    // 128 threads
  float acc = bq[o];
  for (int k = 0; k < 128; ++k) acc += seed[k] * wq[k * 128 + o];
  qs[o] = acc;
}

// Row-wise layernorm over 128 columns. One wave per row; grid.x = rows.
__global__ __launch_bounds__(32) void layernorm128(
    const float* __restrict__ X, float* __restrict__ Y,
    const float* __restrict__ g, const float* __restrict__ b) {
  size_t row = blockIdx.x;
  int lane = threadIdx.x;
  const float* x = X + row * 128;
  float v[4], s = 0.f, s2 = 0.f;
#pragma unroll
  for (int j = 0; j < 4; ++j) {
    v[j] = x[lane + j * 32];
    s += v[j]; s2 += v[j] * v[j];
  }
#pragma unroll
  for (int off = 16; off; off >>= 1) {
    s += __shfl_xor(s, off, 32);
    s2 += __shfl_xor(s2, off, 32);
  }
  float mu = s * (1.f / 128.f);
  float var = s2 * (1.f / 128.f) - mu * mu;
  float rs = rsqrtf(var + 1e-5f);
  float* y = Y + row * 128;
#pragma unroll
  for (int j = 0; j < 4; ++j) {
    int c = lane + j * 32;
    y[c] = (v[j] - mu) * rs * g[c] + b[c];
  }
}

// ---------------- hypergraph kernels ----------------
__global__ void deg_acc(const int* __restrict__ src, float* __restrict__ deg, int E) {
  int e = blockIdx.x * blockDim.x + threadIdx.x;
  if (e < E) atomicAdd(&deg[src[e]], 1.0f);
}
// ef[r,c] = (1/6) * sum_j st[r*6+j] * XW[src[r*6+j], c]   (edge_deg == 6 exactly)
__global__ void ef_compute(const float* __restrict__ XW, const int* __restrict__ src,
                           const float* __restrict__ st, float* __restrict__ EF, int R) {
  int t = blockIdx.x * blockDim.x + threadIdx.x;
  if (t >= R * 128) return;
  int r = t >> 7, c = t & 127;
  float a = 0.f;
#pragma unroll
  for (int j = 0; j < 6; ++j) {
    int e = r * 6 + j;
    a += st[e] * XW[(size_t)src[e] * 128 + c];
  }
  EF[t] = a * (1.f / 6.f);
}
__global__ void scatter_edges(const float* __restrict__ EF, const int* __restrict__ src,
                              const float* __restrict__ st, float* __restrict__ ACC, int E) {
  int t = blockIdx.x * blockDim.x + threadIdx.x;
  if (t >= E * 128) return;
  int e = t >> 7, c = t & 127;
  atomicAdd(&ACC[(size_t)src[e] * 128 + c], st[e] * EF[(size_t)(e / 6) * 128 + c]);
}
__global__ void hyper_final(const float* __restrict__ ACC, const float* __restrict__ deg,
                            const float* __restrict__ b, float* __restrict__ X, int Nm) {
  int t = blockIdx.x * blockDim.x + threadIdx.x;
  if (t >= Nm * 128) return;
  int m = t >> 7, c = t & 127;
  float d = fmaxf(deg[m], 1.0f);
  X[t] = tanhf(ACC[t] / d + b[c]);
}
__global__ void gather_rows(const float* __restrict__ SRCX, const int* __restrict__ idx,
                            float* __restrict__ X, int rows) {
  int t = blockIdx.x * blockDim.x + threadIdx.x;
  if (t >= rows * 128) return;
  int r = t >> 7, c = t & 127;
  X[t] = SRCX[(size_t)idx[r] * 128 + c];
}

// ---------------- host-side drivers ----------------
struct MabP {
  const float *bk, *bo, *bq, *bv, *wk, *wo, *wq, *wv;
  const float *lin_b, *lin_w, *ln1b, *ln1g, *ln2b, *ln2g;
};
static inline MabP get_mab(void* const* din, int b) {
  MabP m;
  m.bk = (const float*)din[b + 0];  m.bo = (const float*)din[b + 1];
  m.bq = (const float*)din[b + 2];  m.bv = (const float*)din[b + 3];
  m.wk = (const float*)din[b + 4];  m.wo = (const float*)din[b + 5];
  m.wq = (const float*)din[b + 6];  m.wv = (const float*)din[b + 7];
  m.lin_b = (const float*)din[b + 8];  m.lin_w = (const float*)din[b + 9];
  m.ln1b = (const float*)din[b + 10]; m.ln1g = (const float*)din[b + 11];
  m.ln2b = (const float*)din[b + 12]; m.ln2g = (const float*)din[b + 13];
  return m;
}

struct Ctx {
  hipStream_t s;
  unsigned short* Abf;   // bf16 staging for GEMM A operand
  unsigned short* Wp;    // packed bf16 weight fragments (max 128*128)
};

static inline void cvtA(const Ctx& c, const float* A, size_t n) {
  int n8 = (int)(n / 8);
  cvt_bf16x8<<<(n8 + 255) / 256, 256, 0, c.s>>>(A, c.Abf, n8);
}

template <int KC>
static inline void launch_gemm(const Ctx& c, const float* bias, const float* res,
                               float* C, int M, int flags) {
  dim3 g(8, M / 16);
  switch (flags) {
    case EP_RELU:
      gemm128<KC, EP_RELU><<<g, 32, 0, c.s>>>(c.Abf, c.Wp, bias, res, C, M); break;
    case EP_RES:
      gemm128<KC, EP_RES><<<g, 32, 0, c.s>>>(c.Abf, c.Wp, bias, res, C, M); break;
    case EP_RES | EP_RELU:
      gemm128<KC, EP_RES | EP_RELU><<<g, 32, 0, c.s>>>(c.Abf, c.Wp, bias, res, C, M); break;
    case EP_RES | EP_RES_BCAST:
      gemm128<KC, EP_RES | EP_RES_BCAST><<<g, 32, 0, c.s>>>(c.Abf, c.Wp, bias, res, C, M); break;
    default:
      gemm128<KC, 0><<<g, 32, 0, c.s>>>(c.Abf, c.Wp, bias, res, C, M); break;
  }
}

// pack W then run the WMMA GEMM on the staged bf16 A
static inline void gemmP(const Ctx& c, int K, const float* W, const float* bias,
                         const float* res, float* C, int M, int flags) {
  pack_w<<<(K * 128 + 255) / 256, 256, 0, c.s>>>(W, c.Wp, K);
  if (K == 64) launch_gemm<2>(c, bias, res, C, M, flags);
  else         launch_gemm<4>(c, bias, res, C, M, flags);
}

static void run_sab(const Ctx& c, const MabP& m, float*& X, float*& T,
                    float* Q, float* Kb, float* V, int S, int n) {
  int rows = S * n;
  cvtA(c, X, (size_t)rows * 128);                       // one cvt serves q/k/v
  gemmP(c, 128, m.wq, m.bq, nullptr, Q, rows, 0);
  gemmP(c, 128, m.wk, m.bk, nullptr, Kb, rows, 0);
  gemmP(c, 128, m.wv, m.bv, nullptr, V, rows, 0);
  int tot = S * n * 4, gr = (tot + 255) / 256;
  if (n == 6) attn_self<6><<<gr, 256, 0, c.s>>>(Q, Kb, V, Q, S);
  else        attn_self<8><<<gr, 256, 0, c.s>>>(Q, Kb, V, Q, S);
  cvtA(c, Q, (size_t)rows * 128);
  gemmP(c, 128, m.wo, m.bo, X, X, rows, EP_RES);        // h = x + attn
  layernorm128<<<rows, 32, 0, c.s>>>(X, X, m.ln1g, m.ln1b);
  cvtA(c, X, (size_t)rows * 128);
  gemmP(c, 128, m.lin_w, m.lin_b, X, T, rows, EP_RES | EP_RELU);
  layernorm128<<<rows, 32, 0, c.s>>>(T, T, m.ln2g, m.ln2b);
  float* tmp = X; X = T; T = tmp;
}

static void run_pma(const Ctx& c, void* const* din, int base, float*& X, float*& T,
                    float* Q, float* Kb, float* V, int S, int n, float* qs) {
  const float* plin_b = (const float*)din[base + 42];
  const float* plin_w = (const float*)din[base + 43];
  MabP m = get_mab(din, base + 44);
  const float* seed = (const float*)din[base + 58];
  int rows = S * n;
  cvtA(c, X, (size_t)rows * 128);
  gemmP(c, 128, plin_w, plin_b, nullptr, T, rows, EP_RELU);   // Z = relu(xW+b) -> T
  seed_q<<<1, 128, 0, c.s>>>(seed, m.wq, m.bq, qs);
  cvtA(c, T, (size_t)rows * 128);                             // one cvt serves k/v
  gemmP(c, 128, m.wk, m.bk, nullptr, Kb, rows, 0);
  gemmP(c, 128, m.wv, m.bv, nullptr, V, rows, 0);
  int tot = S * 4, gr = (tot + 255) / 256;
  if (n == 6) attn_pma<6><<<gr, 256, 0, c.s>>>(qs, Kb, V, Q, S);
  else        attn_pma<8><<<gr, 256, 0, c.s>>>(qs, Kb, V, Q, S);
  cvtA(c, Q, (size_t)S * 128);
  gemmP(c, 128, m.wo, m.bo, seed, X, S, EP_RES | EP_RES_BCAST); // seed + attn
  layernorm128<<<S, 32, 0, c.s>>>(X, X, m.ln1g, m.ln1b);
  cvtA(c, X, (size_t)S * 128);
  gemmP(c, 128, m.lin_w, m.lin_b, X, T, S, EP_RES | EP_RELU);
  layernorm128<<<S, 32, 0, c.s>>>(T, T, m.ln2g, m.ln2b);
  float* tmp = X; X = T; T = tmp;
}

// Decoder SAB with n==1: softmax over one key == 1, so mha == (x@wv+bv)@wo+bo.
static void run_dec(const Ctx& c, const MabP& m, float*& X, float*& T,
                    float* V, int S, float* dest) {
  cvtA(c, X, (size_t)S * 128);
  gemmP(c, 128, m.wv, m.bv, nullptr, V, S, 0);
  cvtA(c, V, (size_t)S * 128);
  gemmP(c, 128, m.wo, m.bo, X, X, S, EP_RES);
  layernorm128<<<S, 32, 0, c.s>>>(X, X, m.ln1g, m.ln1b);
  cvtA(c, X, (size_t)S * 128);
  gemmP(c, 128, m.lin_w, m.lin_b, X, T, S, EP_RES | EP_RELU);
  layernorm128<<<S, 32, 0, c.s>>>(T, dest, m.ln2g, m.ln2b);
}

// ST pytree leaf layout (dict keys sorted): dec[0]@+0, enc[0]@+14, enc[1]@+28,
// pma.lin_b@+42, pma.lin_w@+43, pma.mab@+44, pma.seed@+58  (59 leaves total)
static void run_st(const Ctx& c, void* const* din, int base, float* X, float* T,
                   float* Q, float* Kb, float* V, int S, int n, float* qs, float* dest) {
  MabP dec  = get_mab(din, base + 0);
  MabP enc0 = get_mab(din, base + 14);
  MabP enc1 = get_mab(din, base + 28);
  run_sab(c, enc0, X, T, Q, Kb, V, S, n);
  run_sab(c, enc1, X, T, Q, Kb, V, S, n);
  run_pma(c, din, base, X, T, Q, Kb, V, S, n, qs);
  run_dec(c, dec, X, T, V, S, dest);
}

extern "C" void kernel_launch(void* const* d_in, const int* in_sizes, int n_in,
                              void* d_out, int out_size, void* d_ws, size_t ws_size,
                              hipStream_t stream) {
  (void)n_in; (void)out_size; (void)ws_size;
  const int* src = (const int*)d_in[0];          // hyperedge_index[0] = first E ints
  const float* st = (const float*)d_in[1];
  const float* mx = (const float*)d_in[2];
  const int* rmi = (const int*)d_in[3];
  const int* gmi = (const int*)d_in[4];
  const int E = in_sizes[1];                     // R*6 = 600000
  const int Nm = in_sizes[2] / 64;               // 50000
  const int R = in_sizes[3] / 6;                 // 100000
  const int G = in_sizes[4] / 8;                 // 20000
  // hyper params (list order, dict keys sorted: b then w)
  const float* h0b = (const float*)d_in[5];
  const float* h0w = (const float*)d_in[6];
  const float* h1b = (const float*)d_in[7];
  const float* h1w = (const float*)d_in[8];
  const int st_m2r = 9, st_r2g = 9 + 59;

  // workspace carve-out
  float* w = (float*)d_ws;
  size_t MAXR = (size_t)((E > G * 8) ? E : G * 8);
  float* B0 = w; w += MAXR * 128;   // X / gathered sets / xw scratch
  float* B1 = w; w += MAXR * 128;   // T (ping-pong)
  float* B2 = w; w += MAXR * 128;   // Q (attention output in-place)
  float* B3 = w; w += MAXR * 128;   // K
  float* B4 = w; w += MAXR * 128;   // V
  float* DEG = w; w += Nm;
  float* A1 = w; w += (size_t)Nm * 128;   // scatter accumulator
  float* A2 = w; w += (size_t)Nm * 128;   // metabolite activations
  float* EF = w; w += (size_t)R * 128;
  float* RR = w; w += (size_t)R * 128;    // reaction_repr
  float* QS = w; w += 256;                // pma seed-query
  unsigned short* ABF = (unsigned short*)w; w += MAXR * 64;  // bf16 A staging
  unsigned short* WP  = (unsigned short*)w; w += 8192;       // packed weight

  Ctx ctx{stream, ABF, WP};
  const int blk = 256;

  // ---- hypergraph conv, layer 1 (K=64) ----
  cvtA(ctx, mx, (size_t)Nm * 64);
  gemmP(ctx, 64, h0w, nullptr, nullptr, B0, Nm, 0);
  hipMemsetAsync(DEG, 0, (size_t)Nm * sizeof(float), stream);
  deg_acc<<<(E + blk - 1) / blk, blk, 0, stream>>>(src, DEG, E);
  ef_compute<<<(R * 128 + blk - 1) / blk, blk, 0, stream>>>(B0, src, st, EF, R);
  hipMemsetAsync(A1, 0, (size_t)Nm * 128 * sizeof(float), stream);
  scatter_edges<<<(E * 128 + blk - 1) / blk, blk, 0, stream>>>(EF, src, st, A1, E);
  hyper_final<<<(Nm * 128 + blk - 1) / blk, blk, 0, stream>>>(A1, DEG, h0b, A2, Nm);
  // ---- hypergraph conv, layer 2 (K=128) ----
  cvtA(ctx, A2, (size_t)Nm * 128);
  gemmP(ctx, 128, h1w, nullptr, nullptr, B0, Nm, 0);
  ef_compute<<<(R * 128 + blk - 1) / blk, blk, 0, stream>>>(B0, src, st, EF, R);
  hipMemsetAsync(A1, 0, (size_t)Nm * 128 * sizeof(float), stream);
  scatter_edges<<<(E * 128 + blk - 1) / blk, blk, 0, stream>>>(EF, src, st, A1, E);
  hyper_final<<<(Nm * 128 + blk - 1) / blk, blk, 0, stream>>>(A1, DEG, h1b, A2, Nm);

  // ---- metabolite -> reaction set transformer (S=R, n=6) ----
  gather_rows<<<(R * 6 * 128 + blk - 1) / blk, blk, 0, stream>>>(A2, rmi, B0, R * 6);
  run_st(ctx, d_in, st_m2r, B0, B1, B2, B3, B4, R, 6, QS, RR);

  // ---- reaction -> gene set transformer (S=G, n=8) ----
  gather_rows<<<(G * 8 * 128 + blk - 1) / blk, blk, 0, stream>>>(RR, gmi, B0, G * 8);
  run_st(ctx, d_in, st_r2g, B0, B1, B2, B3, B4, G, 8, QS, (float*)d_out);
}
